// ToUCrossAttention_8186207666960
// MI455X (gfx1250) — matrix operand order repaired and checked by reading, not compiled, not run on hardware
//
#include <hip/hip_runtime.h>

typedef __attribute__((ext_vector_type(8)))  float   v8f;
typedef __attribute__((ext_vector_type(8)))  __bf16  v8bf;
typedef __attribute__((ext_vector_type(16))) __bf16  v16bf;

#define DMm 1024
#define DPp 128
#define PNn 256
#define TMr 32   // tokens per block

struct Frag4 { v16bf f[4]; };

// ---------- helpers ----------
__device__ __forceinline__ __bf16 f2bf(float f) {
  union { float f; unsigned u; } v; v.f = f;
  unsigned r = (v.u + 0x7FFFu + ((v.u >> 16) & 1u)) >> 16;  // RNE
  unsigned short s = (unsigned short)r;
  __bf16 b;
  __builtin_memcpy(&b, &s, 2);
  return b;
}

// A fragment: elems 0..15 at K=k0..k0+15 (contiguous)
__device__ __forceinline__ v16bf ld_fragA(const __bf16* p) {
  v8bf lo = *(const v8bf*)(p);
  v8bf hi = *(const v8bf*)(p + 8);
  return __builtin_shufflevector(lo, hi, 0,1,2,3,4,5,6,7,8,9,10,11,12,13,14,15);
}
// B fragment: 16 contiguous bf16 per lane (K-major row of Wt[n][k])
__device__ __forceinline__ v16bf ld_fragB(const __bf16* p) {
  v8bf lo = *(const v8bf*)(p);
  v8bf hi = *(const v8bf*)(p + 8);
  return __builtin_shufflevector(lo, hi, 0,1,2,3,4,5,6,7,8,9,10,11,12,13,14,15);
}
__device__ __forceinline__ v8f wmma_bf(v16bf a, v16bf b, v8f c) {
  return __builtin_amdgcn_wmma_f32_16x16x32_bf16(false, a, false, b, (short)0, c, false, false);
}

// ---------- prep kernel 1: convert Wq [128][1024] and Wo [1024][128] to bf16 ----------
__global__ __launch_bounds__(256) void prep_conv(const float* __restrict__ Wq,
                                                 const float* __restrict__ Wo,
                                                 __bf16* __restrict__ Wq_bf,
                                                 __bf16* __restrict__ Wo_bf) {
  int i = blockIdx.x * 256 + threadIdx.x;  // 262144 total
  if (i < 131072) Wq_bf[i] = f2bf(Wq[i]);
  else            Wo_bf[i - 131072] = f2bf(Wo[i - 131072]);
}

// ---------- prep kernel 2: k = prim·Wk^T (as [256][128]) and v^T (as [128][256]) ----------
__global__ __launch_bounds__(256) void prep_kv(const float* __restrict__ prim,
                                               const float* __restrict__ Wk,
                                               const float* __restrict__ Wv,
                                               __bf16* __restrict__ k_bf,
                                               __bf16* __restrict__ vT_bf) {
  int idx = blockIdx.x * 256 + threadIdx.x;  // 32768 total
  int p = idx >> 7, d = idx & 127;
  const float* pr = prim + p * DPp;
  const float* wk = Wk + d * DPp;
  const float* wv = Wv + d * DPp;
  float ka = 0.f, va = 0.f;
#pragma unroll 8
  for (int j = 0; j < DPp; ++j) { float pv = pr[j]; ka += pv * wk[j]; va += pv * wv[j]; }
  k_bf[p * DPp + d]  = f2bf(ka);
  vT_bf[d * PNn + p] = f2bf(va);
}

// ---------- fused main kernel: 1 block = 32 tokens, 4 waves ----------
// LDS regions (bytes):
//   [0,16384)      : x_chunk bf16 [32][256]  (reused as attn bf16 [32][256])
//   [16384,24576)  : q bf16 [32][128]
//   [24576,57344)  : scores f32 [32][256]    (reused as ctx bf16 [32][128])
//   [57344,57856)  : red f32 [128]
//   [57856,57984)  : gate f32 [32]
//   [57984,58112)  : rnorm f32 [32]
__global__ __launch_bounds__(128) void fused_attn(const float* __restrict__ x,
                                                  const float* __restrict__ gate_w,
                                                  const float* __restrict__ gate_b,
                                                  const __bf16* __restrict__ Wq_bf,
                                                  const __bf16* __restrict__ Wo_bf,
                                                  const __bf16* __restrict__ k_bf,
                                                  const __bf16* __restrict__ vT_bf,
                                                  float* __restrict__ out) {
  __shared__ alignas(16) char smem[58112];
  __bf16* xch   = (__bf16*)(smem);             // [32][256]
  __bf16* attn  = (__bf16*)(smem);             // [32][256] (after stage 1/2)
  __bf16* q_l   = (__bf16*)(smem + 16384);     // [32][128]
  float*  s_l   = (float*) (smem + 24576);     // [32][256]
  __bf16* ctx_l = (__bf16*)(smem + 24576);     // [32][128] (after softmax)
  float*  red   = (float*) (smem + 57344);     // [128]
  float*  gate  = (float*) (smem + 57856);     // [32]
  float*  rnorm = (float*) (smem + 57984);     // [32]

  const int t     = threadIdx.x;
  const int wave  = t >> 5;
  const int lane  = t & 31;
  const int laneM = lane & 15;     // N/M column within tile
  const int laneH = lane >> 4;     // half-wave select
  const int row0  = blockIdx.x * TMr;
  const int mtile = wave & 1;      // 2 m-tiles of 16 rows
  const int arow  = mtile * 16 + laneM;          // A-fragment source row
  const int drow  = mtile * 16 + laneH * 8;      // D-fragment base row

  // ================= Stage 1: q = x·Wq^T, + gate dot =================
  const int ntb1 = (wave >> 1) * 4;
  v8f qacc[4];
#pragma unroll
  for (int n = 0; n < 4; ++n) qacc[n] = (v8f){0,0,0,0,0,0,0,0};

  const int lrow = t >> 2;            // loader: 4 threads per row
  float gpart = 0.f;
  for (int chunk = 0; chunk < 4; ++chunk) {
    const int kbase = chunk * 256;
    __syncthreads();
    // load 32x256 f32 -> bf16 LDS
#pragma unroll
    for (int i = 0; i < 16; ++i) {
      int j   = (t & 3) + 4 * i;      // float4 index within row (0..63)
      int col = j * 4;
      float4 xv = *(const float4*)(x + (row0 + lrow) * DMm + kbase + col);
      float4 gv = *(const float4*)(gate_w + kbase + col);
      gpart += xv.x * gv.x + xv.y * gv.y + xv.z * gv.z + xv.w * gv.w;
      __bf16* d = xch + lrow * 256 + col;
      d[0] = f2bf(xv.x); d[1] = f2bf(xv.y); d[2] = f2bf(xv.z); d[3] = f2bf(xv.w);
    }
    __syncthreads();

    // software-pipelined: B group (4 frags) prefetched one k-step ahead
    Frag4 bA, bB;
#pragma unroll
    for (int n = 0; n < 4; ++n)
      bA.f[n] = ld_fragB(Wq_bf + ((ntb1 + n) * 16 + laneM) * DMm + kbase + laneH * 16);
    v16bf a = ld_fragA(xch + arow * 256 + laneH * 16);
#pragma unroll
    for (int ks = 0; ks < 8; ++ks) {
      Frag4& cur = (ks & 1) ? bB : bA;
      Frag4& nxt = (ks & 1) ? bA : bB;
      v16bf acur = a;
      if (ks < 7) {
#pragma unroll
        for (int n = 0; n < 4; ++n)
          nxt.f[n] = ld_fragB(Wq_bf + ((ntb1 + n) * 16 + laneM) * DMm + kbase + (ks + 1) * 32 + laneH * 16);
        a = ld_fragA(xch + arow * 256 + (ks + 1) * 32 + laneH * 16);
      }
#pragma unroll
      for (int n = 0; n < 4; ++n) qacc[n] = wmma_bf(acur, cur.f[n], qacc[n]);
    }
  }
  red[t] = gpart;  // deterministic gate reduction (4 partials/row)
#pragma unroll
  for (int n = 0; n < 4; ++n)
#pragma unroll
    for (int r = 0; r < 8; ++r)
      q_l[(drow + r) * DPp + (ntb1 + n) * 16 + laneM] = f2bf(qacc[n][r]);
  __syncthreads();
  if (t < 32) gate[t] = gate_b[0] + red[t*4] + red[t*4+1] + red[t*4+2] + red[t*4+3];

  // ================= Stage 2: scores = q·k^T =================
  // 8 groups: g = ks*2 + h; group g covers ntiles ntb2 + h*4 + {0..3}
  const int ntb2 = (wave >> 1) * 8;
  v8f sacc[8];
#pragma unroll
  for (int n = 0; n < 8; ++n) sacc[n] = (v8f){0,0,0,0,0,0,0,0};
  {
    Frag4 bA, bB;
    v16bf aA, aB;
#pragma unroll
    for (int n = 0; n < 4; ++n)
      bA.f[n] = ld_fragB(k_bf + ((ntb2 + n) * 16 + laneM) * DPp + laneH * 16);
    aA = ld_fragA(q_l + arow * DPp + laneH * 16);
#pragma unroll
    for (int g = 0; g < 8; ++g) {
      const int ks = g >> 1, h = g & 1;
      Frag4& cur = (g & 1) ? bB : bA;
      Frag4& nxt = (g & 1) ? bA : bB;
      v16bf acur = (ks & 1) ? aB : aA;
      if (h == 0 && ks < 3) {   // prefetch A for next k-step (depth 2)
        v16bf an = ld_fragA(q_l + arow * DPp + (ks + 1) * 32 + laneH * 16);
        if ((ks + 1) & 1) aB = an; else aA = an;
      }
      if (g < 7) {
        const int gn = g + 1, ksn = gn >> 1, hn = gn & 1;
#pragma unroll
        for (int n = 0; n < 4; ++n)
          nxt.f[n] = ld_fragB(k_bf + ((ntb2 + hn * 4 + n) * 16 + laneM) * DPp + ksn * 32 + laneH * 16);
      }
#pragma unroll
      for (int n = 0; n < 4; ++n) sacc[h * 4 + n] = wmma_bf(acur, cur.f[n], sacc[h * 4 + n]);
    }
  }
#pragma unroll
  for (int n = 0; n < 8; ++n)
#pragma unroll
    for (int r = 0; r < 8; ++r)
      s_l[(drow + r) * PNn + (ntb2 + n) * 16 + laneM] = sacc[n][r];
  __syncthreads();

  // ================= Stage 3: softmax over 256 (4 threads/row) =================
  const float scale = 0.088388347648318447f;  // 1/sqrt(128)
  const int srow = t >> 2, sp = t & 3;
  float m = -3.4e38f;
#pragma unroll 8
  for (int c = 0; c < 64; ++c) m = fmaxf(m, s_l[srow * PNn + sp * 64 + c]);
  red[t] = m;
  __syncthreads();
  float rmax = fmaxf(fmaxf(red[srow*4], red[srow*4+1]), fmaxf(red[srow*4+2], red[srow*4+3]));
  __syncthreads();
  float ssum = 0.f;
#pragma unroll 8
  for (int c = 0; c < 64; ++c) {
    float e = __expf(scale * (s_l[srow * PNn + sp * 64 + c] - rmax));
    attn[srow * PNn + sp * 64 + c] = f2bf(e);   // unnormalized; 1/sum folded in later
    ssum += e;
  }
  red[t] = ssum;
  __syncthreads();
  if (sp == 0)
    rnorm[srow] = 1.0f / (red[srow*4] + red[srow*4+1] + red[srow*4+2] + red[srow*4+3]);
  __syncthreads();

  // ================= Stage 4: ctx = attn·v (B = v^T) =================
  const int ntb4 = (wave >> 1) * 4;
  v8f cacc[4];
#pragma unroll
  for (int n = 0; n < 4; ++n) cacc[n] = (v8f){0,0,0,0,0,0,0,0};
  {
    Frag4 bA, bB;
#pragma unroll
    for (int n = 0; n < 4; ++n)
      bA.f[n] = ld_fragB(vT_bf + ((ntb4 + n) * 16 + laneM) * PNn + laneH * 16);
    v16bf a = ld_fragA(attn + arow * PNn + laneH * 16);
#pragma unroll
    for (int ks = 0; ks < 8; ++ks) {
      Frag4& cur = (ks & 1) ? bB : bA;
      Frag4& nxt = (ks & 1) ? bA : bB;
      v16bf acur = a;
      if (ks < 7) {
#pragma unroll
        for (int n = 0; n < 4; ++n)
          nxt.f[n] = ld_fragB(vT_bf + ((ntb4 + n) * 16 + laneM) * PNn + (ks + 1) * 32 + laneH * 16);
        a = ld_fragA(attn + arow * PNn + (ks + 1) * 32 + laneH * 16);
      }
#pragma unroll
      for (int n = 0; n < 4; ++n) cacc[n] = wmma_bf(acur, cur.f[n], cacc[n]);
    }
  }
  __syncthreads();   // s_l readers done; safe to overwrite region with ctx
#pragma unroll
  for (int n = 0; n < 4; ++n)
#pragma unroll
    for (int r = 0; r < 8; ++r)
      ctx_l[(drow + r) * DPp + (ntb4 + n) * 16 + laneM] = f2bf(cacc[n][r] * rnorm[drow + r]);
  __syncthreads();

  // ================= Stage 5: out = gate ⊙ (ctx·Wo^T) =================
  v16bf af[4];
#pragma unroll
  for (int ks = 0; ks < 4; ++ks)
    af[ks] = ld_fragA(ctx_l + arow * DPp + ks * 32 + laneH * 16);
  float gsig[8];
#pragma unroll
  for (int r = 0; r < 8; ++r) {
    float g = gate[drow + r];
    gsig[r] = 1.0f / (1.0f + __expf(-g));
  }
  const int ntb5 = (wave >> 1) * 32;
  {
    Frag4 bA, bB;
#pragma unroll
    for (int ks = 0; ks < 4; ++ks)
      bA.f[ks] = ld_fragB(Wo_bf + (ntb5 * 16 + laneM) * DPp + ks * 32 + laneH * 16);
#pragma unroll 2
    for (int n = 0; n < 32; ++n) {
      const int nt = ntb5 + n;
      Frag4& cur = (n & 1) ? bB : bA;
      Frag4& nxt = (n & 1) ? bA : bB;
      if (n < 31) {
#pragma unroll
        for (int ks = 0; ks < 4; ++ks)
          nxt.f[ks] = ld_fragB(Wo_bf + ((nt + 1) * 16 + laneM) * DPp + ks * 32 + laneH * 16);
      }
      v8f oacc = (v8f){0,0,0,0,0,0,0,0};
#pragma unroll
      for (int ks = 0; ks < 4; ++ks) oacc = wmma_bf(af[ks], cur.f[ks], oacc);
#pragma unroll
      for (int r = 0; r < 8; ++r)
        out[(row0 + drow + r) * DMm + nt * 16 + laneM] = gsig[r] * oacc[r];
    }
  }
}

// ---------- launch ----------
extern "C" void kernel_launch(void* const* d_in, const int* in_sizes, int n_in,
                              void* d_out, int out_size, void* d_ws, size_t ws_size,
                              hipStream_t stream) {
  const float* x    = (const float*)d_in[0];
  const float* prim = (const float*)d_in[1];
  const float* Wq   = (const float*)d_in[2];
  const float* Wk   = (const float*)d_in[3];
  const float* Wv   = (const float*)d_in[4];
  const float* Wo   = (const float*)d_in[5];
  const float* gw   = (const float*)d_in[6];
  const float* gb   = (const float*)d_in[7];
  float* out = (float*)d_out;

  char* ws = (char*)d_ws;
  __bf16* Wq_bf = (__bf16*)(ws);             // 128*1024 bf16 = 256 KB
  __bf16* Wo_bf = (__bf16*)(ws + 262144);    // 1024*128 bf16 = 256 KB
  __bf16* k_bf  = (__bf16*)(ws + 524288);    // 256*128 bf16 = 64 KB
  __bf16* vT_bf = (__bf16*)(ws + 589824);    // 128*256 bf16 = 64 KB

  prep_conv<<<1024, 256, 0, stream>>>(Wq, Wo, Wq_bf, Wo_bf);
  prep_kv<<<128, 256, 0, stream>>>(prim, Wk, Wv, k_bf, vT_bf);

  const int nblocks = (8 * 4096) / TMr;      // 1024 blocks of 32 tokens
  fused_attn<<<nblocks, 128, 0, stream>>>(x, gw, gb, Wq_bf, Wo_bf, k_bf, vT_bf, out);
}
